// CrfDecoderABC_26156350833020
// MI455X (gfx1250) — compile-verified, matrix operand order checked by reading, not run on hardware
//
#include <hip/hip_runtime.h>
#include <hip/hip_bf16.h>
#include <stdint.h>

// Problem constants (from reference)
#define TT 1024
#define BB 64
#define CC 8
#define NN 64

typedef __attribute__((ext_vector_type(16))) _Float16 v16h;
typedef __attribute__((ext_vector_type(8)))  _Float16 v8h;
typedef __attribute__((ext_vector_type(8)))  float    v8f;

#define HSTR 80   // f16 row stride (halves): 160B, multiple of 16B
#define ASTR 68   // f32 row stride (floats): 272B, multiple of 16B

#define L2E 1.44269504088896340736f   // log2(e)
#define LN2 0.69314718055994530942f   // ln(2)

// gfx1250 async global->LDS DMA (ASYNCcnt-tracked). Guarded so both host pass
// and toolchains without the builtin fall back to synchronous staging.
#if defined(__has_builtin)
#if __has_builtin(__builtin_amdgcn_global_load_async_to_lds_b128) && \
    __has_builtin(__builtin_amdgcn_s_wait_asynccnt)
#define ASYNC_LDS 1
#endif
#endif
#ifndef ASYNC_LDS
#define ASYNC_LDS 0
#endif

#if ASYNC_LDS
// Builtin signature (from compiler diagnostic): parameters are pointers to
// 16B int vectors: (global int4*, shared int4*, imm offset, imm cpol).
typedef int v4i __attribute__((vector_size(16)));
typedef void __attribute__((address_space(1))) gas_void;
typedef void __attribute__((address_space(3))) las_void;
typedef v4i  __attribute__((address_space(1))) *gv4ip;
typedef v4i  __attribute__((address_space(3))) *lv4ip;
#define ASYNC_B128(gsrc, ldst, off) \
    __builtin_amdgcn_global_load_async_to_lds_b128( \
        (gv4ip)(gas_void*)(gsrc), (lv4ip)(las_void*)(ldst), (off), 0)
#endif

// 16-bit WMMA A/B fragment per ISA layout:
//   lane<16: K = kb + {0..7, 16..23};  lane>=16: K = kb + {8..15, 24..31}
// Row-major storage -> two contiguous 16B runs -> ds_read_b128 x2.
__device__ __forceinline__ v16h load_frag_f16(const _Float16* rowbase, int kb, int half8) {
    const v8h* p0 = reinterpret_cast<const v8h*>(rowbase + kb + half8);
    const v8h* p1 = reinterpret_cast<const v8h*>(rowbase + kb + 16 + half8);
    v8h lo = *p0;
    v8h hi = *p1;
    v16h r;
#pragma unroll
    for (int e = 0; e < 8; ++e) { r[e] = lo[e]; r[e + 8] = hi[e]; }
    return r;
}

// ---------------------------------------------------------------------------
// Kernel 1: path score (em_total + head + tail + trans_total) -> d_ws [B*C]
// ---------------------------------------------------------------------------
__global__ void crf_scores_kernel(const float* __restrict__ em,
                                  const int*   __restrict__ tags,
                                  const int*   __restrict__ lengths,
                                  const float* __restrict__ trans,
                                  const float* __restrict__ head,
                                  const float* __restrict__ tail,
                                  float* __restrict__ scores) {
    int idx = blockIdx.x * blockDim.x + threadIdx.x;   // [0, B*C)
    if (idx >= BB * CC) return;
    int b = idx >> 3;
    int c = idx & 7;
    int len = lengths[b];

    int tag0 = tags[(0 * BB + b) * CC + c];
    float tot = em[(((size_t)0 * BB + b) * CC + c) * NN + tag0];
    float hs  = head[c * NN + tag0];
    int prev = tag0;

    for (int t = 1; t < TT; ++t) {
        int tag = tags[((size_t)t * BB + b) * CC + c];
        if (t < len) {
            tot += em[(((size_t)t * BB + b) * CC + c) * NN + tag];
            tot += trans[((size_t)c * NN + prev) * NN + tag];
        }
        prev = tag;
    }
    int tagL = tags[((size_t)(len - 1) * BB + b) * CC + c];
    scores[b * CC + c] = tot + hs + tail[c * NN + tagL];
}

// ---------------------------------------------------------------------------
// Kernel 2: WMMA forward recursion + log-partition; out = scores - logZ
// One block per conjugate c; 256 threads = 8 wave32s; 2 barriers per step.
// ---------------------------------------------------------------------------
__global__ void __launch_bounds__(256)
crf_forward_kernel(const float* __restrict__ em,
                   const int*   __restrict__ lengths,
                   const float* __restrict__ trans,
                   const float* __restrict__ head,
                   const float* __restrict__ tail,
                   const float* __restrict__ scores,
                   float* __restrict__ out) {
    __shared__ float    sAlpha[BB * ASTR];     // alpha[b][i], f32
    __shared__ float    sEm   [BB * ASTR];     // emissions[t][b][i], staged
    __shared__ _Float16 sP [BB * HSTR];        // exp(alpha - m) as f16, [b][k]
    __shared__ _Float16 sET[NN * HSTR];        // E^T[j][k] = exp(trans[c][k][j]) f16
    __shared__ float    sM[BB];                // per-row max

    const int c     = blockIdx.x;              // 0..7
    const int tid   = threadIdx.x;
    const int lane  = tid & 31;
    const int w     = tid >> 5;                // wave id 0..7
    const int half8 = (lane >> 4) << 3;
    const int l15   = lane & 15;

    // Static tile ownership: wave w owns tiles {w, w+8}; both share tn = w&3.
    const int tn = w & 3;
    const int j  = tn * 16 + l15;              // output/state column (fixed)

    // --- init: E^T (constant per block) and alpha0 = head + emissions[0] ---
    for (int e = tid; e < NN * NN; e += 256) {
        int jj = e >> 6, k = e & 63;
        sET[jj * HSTR + k] =
            (_Float16)__builtin_amdgcn_exp2f(trans[((size_t)c * NN + k) * NN + jj] * L2E);
    }
    for (int e = tid; e < BB * NN; e += 256) {
        int b = e >> 6, i = e & 63;
        sAlpha[b * ASTR + i] =
            head[c * NN + i] + em[(((size_t)0 * BB + b) * CC + c) * NN + i];
    }

    // Per-thread register state for the 16 alpha slots this thread owns.
    int   lenr[2][8];
    float aold[2][8];
#pragma unroll
    for (int tp = 0; tp < 2; ++tp) {
        const int tm = (tp << 1) + (w >> 2);   // tp=0 -> tm in {0,1}; tp=1 -> {2,3}
        const int rb = tm * 16 + half8;        // base batch row
#pragma unroll
        for (int r = 0; r < 8; ++r) lenr[tp][r] = lengths[rb + r];
    }
    __syncthreads();
#pragma unroll
    for (int tp = 0; tp < 2; ++tp) {
        const int tm = (tp << 1) + (w >> 2);
        const int rb = tm * 16 + half8;
#pragma unroll
        for (int r = 0; r < 8; ++r) aold[tp][r] = sAlpha[(rb + r) * ASTR + j];
    }

    // --- sequential forward recursion over time ---
    for (int t = 1; t < TT; ++t) {
        // phase A: kick off em[t] DMA, row max (4 threads/row + shfl_xor),
        //          exp -> f16; wait DMA; barrier.
        {
            const int b  = tid >> 2;           // 0..63
            const int i0 = (tid & 3) * 16;

            // stage emissions[t, b, c, i0..i0+15] -> sEm
            const float* gsrc = &em[(((size_t)t * BB + b) * CC + c) * NN + i0];
            float*       ldst = &sEm[b * ASTR + i0];
#if ASYNC_LDS
            // 4 x 16B async DMA; same literal offset advances global & LDS addr.
            ASYNC_B128(gsrc, ldst, 0);
            ASYNC_B128(gsrc, ldst, 16);
            ASYNC_B128(gsrc, ldst, 32);
            ASYNC_B128(gsrc, ldst, 48);
#endif

            const float4* pa = reinterpret_cast<const float4*>(&sAlpha[b * ASTR + i0]);
            float4 a0 = pa[0], a1 = pa[1], a2 = pa[2], a3 = pa[3];
            float mm = fmaxf(fmaxf(fmaxf(a0.x, a0.y), fmaxf(a0.z, a0.w)),
                             fmaxf(fmaxf(a1.x, a1.y), fmaxf(a1.z, a1.w)));
            mm = fmaxf(mm, fmaxf(fmaxf(fmaxf(a2.x, a2.y), fmaxf(a2.z, a2.w)),
                                 fmaxf(fmaxf(a3.x, a3.y), fmaxf(a3.z, a3.w))));
            mm = fmaxf(mm, __shfl_xor(mm, 1, 32));
            mm = fmaxf(mm, __shfl_xor(mm, 2, 32));
            if ((tid & 3) == 0) sM[b] = mm;

            _Float16* pp = &sP[b * HSTR + i0];
            const float av[16] = { a0.x,a0.y,a0.z,a0.w, a1.x,a1.y,a1.z,a1.w,
                                   a2.x,a2.y,a2.z,a2.w, a3.x,a3.y,a3.z,a3.w };
#pragma unroll
            for (int i = 0; i < 16; ++i)
                pp[i] = (_Float16)__builtin_amdgcn_exp2f((av[i] - mm) * L2E);

#if ASYNC_LDS
            __builtin_amdgcn_s_wait_asynccnt(0);   // DMA visible before barrier
#else
            {
                const float4* ge = reinterpret_cast<const float4*>(gsrc);
                float4* se = reinterpret_cast<float4*>(ldst);
#pragma unroll
                for (int q = 0; q < 4; ++q) se[q] = ge[q];
            }
            if (t + 1 < TT)
                __builtin_prefetch(
                    &em[(((size_t)(t + 1) * BB + (tid & 63)) * CC + c) * NN], 0, 0);
#endif
        }
        __syncthreads();

        // phase B: 64x64 = P @ E via v_wmma_f32_16x16x32_f16; straight-line
        // write-back (mask applied with arithmetic blend, no branches).
        {
            // B fragments shared by both owned tiles (same tn).
            const v16h bf0 = load_frag_f16(&sET[j * HSTR], 0,  half8);
            const v16h bf1 = load_frag_f16(&sET[j * HSTR], 32, half8);
#pragma unroll
            for (int tp = 0; tp < 2; ++tp) {
                const int tm   = (tp << 1) + (w >> 2);
                const int arow = tm * 16 + l15;
                const int rb   = tm * 16 + half8;

                v8f acc = {};
                v16h af0 = load_frag_f16(&sP[arow * HSTR], 0,  half8);
                acc = __builtin_amdgcn_wmma_f32_16x16x32_f16(
                    false, af0, false, bf0, (short)0, acc, false, false);
                v16h af1 = load_frag_f16(&sP[arow * HSTR], 32, half8);
                acc = __builtin_amdgcn_wmma_f32_16x16x32_f16(
                    false, af1, false, bf1, (short)0, acc, false, false);

                // per-row max values: 8 contiguous floats -> wide ds reads
                const float4* pm = reinterpret_cast<const float4*>(&sM[rb]);
                float4 m0 = pm[0], m1 = pm[1];
                const float mrow[8] = { m0.x,m0.y,m0.z,m0.w, m1.x,m1.y,m1.z,m1.w };

                // hoist emission column reads so nothing can be sunk under a mask
                float emv[8];
#pragma unroll
                for (int r = 0; r < 8; ++r) emv[r] = sEm[(rb + r) * ASTR + j];

#pragma unroll
                for (int r = 0; r < 8; ++r) {
                    float vnew = __builtin_amdgcn_logf(fmaxf(acc[r], 1e-37f)) * LN2
                               + mrow[r] + emv[r];
                    float msk = (t < lenr[tp][r]) ? 1.0f : 0.0f;
                    float v = __builtin_fmaf(msk, vnew - aold[tp][r], aold[tp][r]);
                    aold[tp][r] = v;
                    sAlpha[(rb + r) * ASTR + j] = v;
                }
            }
        }
        __syncthreads();
    }

    // --- log partition: logsumexp_j(alpha[b][j] + tail[c][j]) ---
    if (tid < BB) {
        const int b = tid;
        float mm = -3.4e38f;
#pragma unroll 8
        for (int jj = 0; jj < NN; ++jj)
            mm = fmaxf(mm, sAlpha[b * ASTR + jj] + tail[c * NN + jj]);
        float s = 0.f;
#pragma unroll 8
        for (int jj = 0; jj < NN; ++jj)
            s += __builtin_amdgcn_exp2f(
                (sAlpha[b * ASTR + jj] + tail[c * NN + jj] - mm) * L2E);
        float logZ = __builtin_amdgcn_logf(s) * LN2 + mm;
        out[b * CC + c] = scores[b * CC + c] - logZ;
    }
}

// ---------------------------------------------------------------------------
extern "C" void kernel_launch(void* const* d_in, const int* in_sizes, int n_in,
                              void* d_out, int out_size, void* d_ws, size_t ws_size,
                              hipStream_t stream) {
    (void)in_sizes; (void)n_in; (void)out_size; (void)ws_size;
    const float* emissions = (const float*)d_in[0];   // [T,B,C,N] f32
    const int*   tags      = (const int*)  d_in[1];   // [T,B,C]
    const int*   lengths   = (const int*)  d_in[2];   // [B]
    const float* trans     = (const float*)d_in[3];   // [C,N,N]
    const float* head      = (const float*)d_in[4];   // [C,N]
    const float* tail      = (const float*)d_in[5];   // [C,N]
    float*       out       = (float*)d_out;           // [B,C]
    float*       scores    = (float*)d_ws;            // B*C floats of scratch

    crf_scores_kernel<<<dim3((BB * CC + 255) / 256), dim3(256), 0, stream>>>(
        emissions, tags, lengths, trans, head, tail, scores);

    crf_forward_kernel<<<dim3(CC), dim3(256), 0, stream>>>(
        emissions, lengths, trans, head, tail, scores, out);
}
